// UNetSsnNet_76647986364464
// MI455X (gfx1250) — compile-verified, matrix operand order checked by reading, not run on hardware
//
#include <hip/hip_runtime.h>
#include <hip/hip_bf16.h>

typedef __attribute__((ext_vector_type(16))) _Float16 v16h;
typedef __attribute__((ext_vector_type(8)))  float    v8f;

#define BATCH 8

// ---------------- workspace layout (bytes) ----------------
#define OFF_E1    0ULL               // 8*6*512*512*4  = 50331648
#define OFF_TMP   50331648ULL        // 50331648 (block conv1 temporaries, max 6ch@512^2)
#define OFF_E2    100663296ULL       // 25165824
#define OFF_P1    125829120ULL       // 12582912
#define OFF_P2    138412032ULL       // 6291456
#define OFF_BT    144703488ULL       // 12582912
#define OFF_CAT2  157286400ULL       // 50331648
#define OFF_D2    207618048ULL       // 25165824
#define OFF_CAT1  100663296ULL       // 100663296 (reuses E2/P1/P2/BT/CAT2, all dead)
#define OFF_D1    201326592ULL       // 50331648  (reuses D2 region, dead)
#define OFF_PIX   0ULL               // 67108864  (reuses E1/TMP, dead)
#define OFF_CEN0  251658240ULL       // 1048576
#define OFF_CEN1  252706816ULL       // 1048576
#define OFF_ST    253755392ULL       // 1024 (per-layer BN sums)
#define OFF_WP    253756416ULL       // packed f16 weights (~49 KB)
#define WS_NEED   253805568ULL

// K enumeration (shared by pack + conv): entry k -> ci = k % Cin; t = k / Cin;
// cc = t % 3; rr = t / 3.  With LDS tile layout [rr][col][ci] this makes
// addr(k) = n*CIN + k + (k/(3*CIN))*15*CIN  — contiguous within each rr row.

// ---------- pack conv weights (f32 OIHW) into WMMA A-fragment layout ----------
__global__ void k_pack_w(const float* __restrict__ w, _Float16* __restrict__ dst,
                         int Cin, int Cout, int mtiles, int ksteps, int total) {
    int idx = blockIdx.x * blockDim.x + threadIdx.x;
    if (idx >= total) return;
    int e    = idx & 15;
    int lane = (idx >> 4) & 31;
    int rest = idx >> 9;            // mt*ksteps + ks
    int ks   = rest % ksteps;
    int mt   = rest / ksteps;
    int m = lane & 15, h = lane >> 4;
    int slot = (e & 7) + 8 * h + 16 * (e >> 3);
    int k  = ks * 32 + slot;
    int co = mt * 16 + m;
    float v = 0.f;
    if (co < Cout && k < Cin * 9) {
        int ci = k % Cin, t = k / Cin, cc = t % 3, rr = t / 3;
        v = w[(co * Cin + ci) * 9 + rr * 3 + cc];
    }
    dst[idx] = (_Float16)v;
}

// ---------- 3x3 conv (edge padding) via implicit GEMM on v_wmma_f32_16x16x32_f16 ----------
// grid: (W/16, H, B), block: 32 (one wave = 16(M)x16(N) tile). MT via template.
template <int CIN, int COUT, int MT, int H, int W>
__global__ __launch_bounds__(32)
void k_conv3x3(const float* __restrict__ in, const _Float16* __restrict__ wp,
               float* __restrict__ out) {
    constexpr int KTOT   = CIN * 9;
    constexpr int KSTEPS = (KTOT + 31) / 32;
    constexpr int HW     = H * W;
    // +8 pad: merged loads of don't-care tail slots must stay in bounds
    __shared__ __align__(16) _Float16 sm[3 * 18 * CIN + 8];

    struct M {
        static __device__ constexpr int off(int k) {
            int kk = (k < CIN * 9) ? k : 0;
            return kk + (kk / (3 * CIN)) * 15 * CIN;
        }
        // in-range elements of [k, k+len) are address-contiguous (invalid = don't care)
        static __device__ constexpr bool contig(int k, int len) {
            if (k >= CIN * 9) return true;
            for (int i = 1; i < len; ++i)
                if (k + i < CIN * 9 && off(k + i) != off(k) + i) return false;
            return true;
        }
    };

    const int x0  = blockIdx.x * 16;
    const int y   = blockIdx.y;
    const int b   = blockIdx.z;
    const int tid = threadIdx.x;

    // cooperative halo load (col fastest => coalesced); edge clamp = jnp 'edge' pad
    constexpr int NEL   = 3 * 18 * CIN;
    constexpr int NFULL = NEL & ~31;
    auto fill = [&](int e) {
        int col = e % 18, t = e / 18, rr = t % 3, ci = t / 3;
        int gy = y + rr - 1;   gy = gy < 0 ? 0 : (gy >= H ? H - 1 : gy);
        int gx = x0 + col - 1; gx = gx < 0 ? 0 : (gx >= W ? W - 1 : gx);
        sm[(rr * 18 + col) * CIN + ci] =
            (_Float16)in[((size_t)(b * CIN + ci)) * HW + gy * W + gx];
    };
#pragma unroll
    for (int e0 = 0; e0 < NFULL; e0 += 32) fill(e0 + tid);
    if (NEL & 31) { if (tid < (NEL & 31)) fill(NFULL + tid); }
    __syncthreads();

    const int n = tid & 15, hh = tid >> 4;
    const int nbase = n * CIN;
    v8f acc = {0.f, 0.f, 0.f, 0.f, 0.f, 0.f, 0.f, 0.f};

    union UBF { v16h h; uint u[8]; };

#pragma unroll
    for (int ks = 0; ks < KSTEPS; ++ks) {
        v16h a = *(const v16h*)(wp + ((size_t)(MT * KSTEPS + ks) * 32 + tid) * 16);
        UBF bf;
#pragma unroll
        for (int g = 0; g < 4; ++g) {            // elements e = 4g .. 4g+3
            const int kg = ks * 32 + 4 * (g & 1) + 16 * (g >> 1);   // k at e=4g, hh=0
            if (CIN % 4 == 0 && M::contig(kg, 4) && M::contig(kg + 8, 4)) {
                // one ds_load_b64 covers 4 f16 elements
                const int A0 = M::off(kg), A1 = M::off(kg + 8);
                uint2 q = *(const uint2*)(sm + nbase + (hh ? A1 : A0));
                bf.u[2 * g]     = q.x;
                bf.u[2 * g + 1] = q.y;
            } else {
#pragma unroll
                for (int p = 0; p < 2; ++p) {    // pairs of elements
                    const int kp = kg + 2 * p;
                    if (CIN % 2 == 0 && M::contig(kp, 2) && M::contig(kp + 8, 2)) {
                        const int B0 = M::off(kp), B1 = M::off(kp + 8);
                        bf.u[2 * g + p] = *(const uint*)(sm + nbase + (hh ? B1 : B0));
                    } else {
#pragma unroll
                        for (int j = 0; j < 2; ++j) {
                            const int ke = kp + j;
                            const int C0 = M::off(ke), C1 = M::off(ke + 8);
                            bf.h[4 * g + 2 * p + j] = sm[nbase + (hh ? C1 : C0)];
                        }
                    }
                }
            }
        }
        acc = __builtin_amdgcn_wmma_f32_16x16x32_f16(false, a, false, bf.h,
                                                     (short)0, acc, false, false);
    }

    // D layout: lane l -> n = l%16, row m = v + 8*(l/16); channel co = MT*16 + m
    const size_t base = ((size_t)b * COUT + MT * 16 + 8 * hh) * HW + (size_t)y * W + x0 + n;
#pragma unroll
    for (int v = 0; v < 8; ++v)
        if (MT * 16 + v + 8 < COUT)                       // both halves valid
            out[base + (size_t)v * HW] = acc[v];
    if (hh == 0) {                                        // single masked block
#pragma unroll
        for (int v = 0; v < 8; ++v)
            if (MT * 16 + v < COUT && MT * 16 + v + 8 >= COUT)
                out[base + (size_t)v * HW] = acc[v];
    }
}

// ---------- per-channel sum / sumsq (one block per channel) ----------
__global__ void k_stats(const float* __restrict__ y, float* __restrict__ st,
                        int C, int HW) {
    __shared__ float s1[256], s2[256];
    int c = blockIdx.x, tid = threadIdx.x;
    float a = 0.f, q = 0.f;
    for (int b = 0; b < BATCH; ++b) {
        const float* p = y + ((size_t)(b * C + c)) * HW;
        for (int i = tid; i < HW; i += 256) { float v = p[i]; a += v; q += v * v; }
    }
    s1[tid] = a; s2[tid] = q; __syncthreads();
    for (int off = 128; off > 0; off >>= 1) {
        if (tid < off) { s1[tid] += s1[tid + off]; s2[tid] += s2[tid + off]; }
        __syncthreads();
    }
    if (tid == 0) { st[c] = s1[0]; st[32 + c] = s2[0]; }
}

// grid (HW/256, C, B)
__global__ void k_bnrelu(float* __restrict__ y, const float* __restrict__ st,
                         const float* __restrict__ g, const float* __restrict__ bb,
                         int C, int HW, int N) {
    int i = blockIdx.x * 256 + threadIdx.x;
    int c = blockIdx.y, b = blockIdx.z;
    float mean = st[c] / (float)N;
    float var  = st[32 + c] / (float)N - mean * mean;
    float sc   = g[c] * rsqrtf(var + 1e-5f);
    size_t o = ((size_t)(b * C + c)) * HW + i;
    y[o] = fmaxf(0.f, (y[o] - mean) * sc + bb[c]);
}

// grid (HWout/256, C, B); sWo = log2(W/2)
__global__ void k_pool(const float* __restrict__ in, float* __restrict__ out,
                       int C, int sWo, int W, int HWin, int HWout) {
    int i = blockIdx.x * 256 + threadIdx.x;
    int c = blockIdx.y, b = blockIdx.z;
    int w = i & ((1 << sWo) - 1), h = i >> sWo;
    const float* p = in + ((size_t)(b * C + c)) * HWin + (2 * h) * W + 2 * w;
    out[((size_t)(b * C + c)) * HWout + i] = fmaxf(fmaxf(p[0], p[1]), fmaxf(p[W], p[W + 1]));
}

// grid (HWout/256, Cout, B); sW2 = log2(2*W)
template <int CIN>
__global__ void k_upconv(const float* __restrict__ x, const float* __restrict__ wt,
                         const float* __restrict__ bias, float* __restrict__ out,
                         int Cout, int sW2, int W, int HWin, int Ctot, int coff, int HWout) {
    int i = blockIdx.x * 256 + threadIdx.x;
    int o = blockIdx.y, b = blockIdx.z;
    int xx = i & ((1 << sW2) - 1), yy = i >> sW2;
    int h = yy >> 1, kh = yy & 1, w = xx >> 1, kw = xx & 1;
    float s = bias[o];
#pragma unroll
    for (int c = 0; c < CIN; ++c)
        s += x[((size_t)(b * CIN + c)) * HWin + h * W + w] * wt[((c * Cout + o) * 2 + kh) * 2 + kw];
    out[((size_t)(b * Ctot + coff + o)) * HWout + i] = s;
}

// grid (HW/256, C, B)
__global__ void k_copych(const float* __restrict__ src, float* __restrict__ dst,
                         int C, int HW, int Ctot, int coff) {
    int i = blockIdx.x * 256 + threadIdx.x;
    int c = blockIdx.y, b = blockIdx.z;
    dst[((size_t)(b * Ctot + coff + c)) * HW + i] = src[((size_t)(b * C + c)) * HW + i];
}

// grid (262144/256, 8, B): ch 0-2 = 1x1 out-conv(d1)+bias, 3-5 = x, 6-7 = coords
__global__ void k_buildpix(const float* __restrict__ d1, const float* __restrict__ ow,
                           const float* __restrict__ ob, const float* __restrict__ x,
                           float* __restrict__ pix) {
    int r = blockIdx.x * 256 + threadIdx.x;     // pixel in 512*512
    int c = blockIdx.y, b = blockIdx.z;
    float v;
    if (c < 3) {
        v = ob[c];
#pragma unroll
        for (int k = 0; k < 6; ++k)
            v += d1[((size_t)(b * 6 + k)) * 262144 + r] * ow[c * 6 + k];
    } else if (c < 6) {
        v = x[((size_t)(b * 3 + (c - 3))) * 262144 + r];
    } else if (c == 6) {
        v = (float)(r >> 9) * 0.0125f;          // y * (M/sh) = y * 0.1/8
    } else {
        v = (float)(r & 511) * 0.0125f;
    }
    pix[((size_t)(b * 8 + c)) * 262144 + r] = v;
}

// grid (16, 8, B): centers = 8x8 average pool of pix
__global__ void k_initc(const float* __restrict__ pix, float* __restrict__ cen) {
    int i = blockIdx.x * 256 + threadIdx.x;     // 0..4095
    int c = blockIdx.y, b = blockIdx.z;
    int nj = i & 63, ni = i >> 6;
    const float* p = pix + ((size_t)(b * 8 + c) * 512 + ni * 8) * 512 + nj * 8;
    float s = 0.f;
#pragma unroll
    for (int r = 0; r < 8; ++r)
#pragma unroll
        for (int q = 0; q < 8; ++q) s += p[r * 512 + q];
    cen[((size_t)(b * 8 + c)) * 4096 + i] = s * (1.f / 64.f);
}

// one block (64 threads) per 8x8 cell; 3x3 neighbor centers cached in LDS
__global__ __launch_bounds__(64)
void k_assign(const float* __restrict__ pix, const float* __restrict__ cen,
              float* __restrict__ sims) {
    __shared__ float sc[9][8];
    __shared__ int   sv[9];
    int cx = blockIdx.x, cy = blockIdx.y, b = blockIdx.z;
    int tid = threadIdx.x;
    for (int t = tid; t < 72; t += 64) {
        int o = t >> 3, c = t & 7;
        int di = o / 3 - 1, dj = o % 3 - 1;
        int ni = cy + di, nj = cx + dj;
        bool valid = (ni >= 0 && ni < 64 && nj >= 0 && nj < 64);
        sc[o][c] = valid ? cen[((size_t)(b * 8 + c) * 64 + ni) * 64 + nj] : 0.f;
        if (c == 0) sv[o] = valid ? 1 : 0;
    }
    __syncthreads();
    int py = cy * 8 + (tid >> 3), px = cx * 8 + (tid & 7);
    float p[8];
#pragma unroll
    for (int c = 0; c < 8; ++c)
        p[c] = pix[((size_t)(b * 8 + c) * 512 + py) * 512 + px];
    float lg[9], mx = -1e30f;
#pragma unroll
    for (int o = 0; o < 9; ++o) {
        if (sv[o]) {
            float d = 0.f;
#pragma unroll
            for (int c = 0; c < 8; ++c) { float t = p[c] - sc[o][c]; d += t * t; }
            lg[o] = -d;                       // sm_scale = 1.0
            if (lg[o] > mx) mx = lg[o];
        } else lg[o] = -1e30f;
    }
    float se = 0.f, ex[9];
#pragma unroll
    for (int o = 0; o < 9; ++o) { float e = sv[o] ? __expf(lg[o] - mx) : 0.f; ex[o] = e; se += e; }
    float inv = 1.f / se;
#pragma unroll
    for (int o = 0; o < 9; ++o)
        sims[((size_t)(b * 9 + o) * 512 + py) * 512 + px] = ex[o] * inv;
}

// one thread per center: gather sims*pix over 9 neighbor cells (= reference scatter-add)
__global__ void k_update(const float* __restrict__ pix, const float* __restrict__ sims,
                         float* __restrict__ cen, int total) {
    int idx = blockIdx.x * blockDim.x + threadIdx.x;
    if (idx >= total) return;
    int nJ = idx & 63, nI = (idx >> 6) & 63, b = idx >> 12;
    float acc[8] = {0, 0, 0, 0, 0, 0, 0, 0};
    float wacc = 0.f;
    for (int o = 0; o < 9; ++o) {
        int di = o / 3 - 1, dj = o % 3 - 1;
        int sy = nI - di, sx = nJ - dj;
        if (sy < 0 || sy >= 64 || sx < 0 || sx >= 64) continue;
        for (int r = 0; r < 8; ++r)
            for (int q = 0; q < 8; ++q) {
                int py = sy * 8 + r, px = sx * 8 + q;
                float s = sims[((size_t)(b * 9 + o) * 512 + py) * 512 + px];
                wacc += s;
#pragma unroll
                for (int c = 0; c < 8; ++c)
                    acc[c] += s * pix[((size_t)(b * 8 + c) * 512 + py) * 512 + px];
            }
    }
    float inv = 1.f / (wacc + 1e-8f);
#pragma unroll
    for (int c = 0; c < 8; ++c)
        cen[((size_t)(b * 8 + c) * 64 + nI) * 64 + nJ] = acc[c] * inv;
}

// ------------------------------ host orchestration ------------------------------
extern "C" void kernel_launch(void* const* d_in, const int* in_sizes, int n_in,
                              void* d_out, int out_size, void* d_ws, size_t ws_size,
                              hipStream_t stream) {
    (void)in_sizes; (void)n_in; (void)out_size;
    if (ws_size < WS_NEED) return;

    const float* x = (const float*)d_in[0];
    char* ws = (char*)d_ws;
    float* E1   = (float*)(ws + OFF_E1);
    float* TMP  = (float*)(ws + OFF_TMP);
    float* E2   = (float*)(ws + OFF_E2);
    float* P1   = (float*)(ws + OFF_P1);
    float* P2   = (float*)(ws + OFF_P2);
    float* BT   = (float*)(ws + OFF_BT);
    float* CAT2 = (float*)(ws + OFF_CAT2);
    float* D2   = (float*)(ws + OFF_D2);
    float* CAT1 = (float*)(ws + OFF_CAT1);
    float* D1   = (float*)(ws + OFF_D1);
    float* PIX  = (float*)(ws + OFF_PIX);
    float* CEN0 = (float*)(ws + OFF_CEN0);
    float* CEN1 = (float*)(ws + OFF_CEN1);
    float* ST   = (float*)(ws + OFF_ST);
    _Float16* WP = (_Float16*)(ws + OFF_WP);
    float* SIMS = (float*)d_out;

    // conv layers: {d_in index, Cin, Cout}
    const int layW[10]   = {1, 4, 7, 10, 13, 16, 21, 24, 29, 32};
    const int layCin[10] = {3, 6, 6, 12, 12, 24, 24, 12, 12, 6};
    const int layCout[10]= {6, 6, 12, 12, 24, 24, 12, 12, 6, 6};
    size_t woff[10]; size_t wcur = 0;
    for (int i = 0; i < 10; ++i) {
        int ksteps = (layCin[i] * 9 + 31) >> 5;
        int mt     = (layCout[i] + 15) >> 4;
        woff[i] = wcur;
        int total = mt * ksteps * 512;
        k_pack_w<<<(total + 255) / 256, 256, 0, stream>>>(
            (const float*)d_in[layW[i]], WP + woff[i], layCin[i], layCout[i], mt, ksteps, total);
        wcur += (size_t)total;
    }

    auto do_bn = [&](float* y, int li, int which, int HW) {
        int C = layCout[li];
        const float* g_ = (const float*)d_in[layW[li] + 1 + which * 3];
        const float* b_ = (const float*)d_in[layW[li] + 2 + which * 3];
        k_stats<<<C, 256, 0, stream>>>(y, ST, C, HW);
        k_bnrelu<<<dim3(HW / 256, C, BATCH), 256, 0, stream>>>(y, ST, g_, b_, C, HW, BATCH * HW);
    };

#define CONV(CI, CO, HH, WW, inp, outp, li)                                              \
    do {                                                                                 \
        k_conv3x3<CI, CO, 0, HH, WW><<<dim3(WW/16, HH, BATCH), 32, 0, stream>>>(         \
            inp, WP + woff[li], outp);                                                   \
        if ((CO) > 16)                                                                   \
            k_conv3x3<CI, CO, ((CO) > 16 ? 1 : 0), HH, WW>                               \
                <<<dim3(WW/16, HH, BATCH), 32, 0, stream>>>(inp, WP + woff[li], outp);   \
    } while (0)

    // ---- UNet ----
    CONV(3, 6, 512, 512, x,   TMP, 0);  do_bn(TMP, 0, 0, 262144);       // enc1 c1
    CONV(6, 6, 512, 512, TMP, E1,  1);  do_bn(E1,  1, 1, 262144);       // enc1 c2
    k_pool<<<dim3(256, 6, BATCH), 256, 0, stream>>>(E1, P1, 6, 8, 512, 262144, 65536);
    CONV(6, 12, 256, 256, P1,  TMP, 2); do_bn(TMP, 2, 0, 65536);        // enc2 c1
    CONV(12,12, 256, 256, TMP, E2,  3); do_bn(E2,  3, 1, 65536);        // enc2 c2
    k_pool<<<dim3(64, 12, BATCH), 256, 0, stream>>>(E2, P2, 12, 7, 256, 65536, 16384);
    CONV(12,24, 128, 128, P2,  TMP, 4); do_bn(TMP, 4, 0, 16384);        // bott c1
    CONV(24,24, 128, 128, TMP, BT,  5); do_bn(BT,  5, 1, 16384);        // bott c2

    k_upconv<24><<<dim3(256, 12, BATCH), 256, 0, stream>>>(              // up2 -> CAT2[0:12)
        BT, (const float*)d_in[19], (const float*)d_in[20], CAT2, 12, 8, 128, 16384, 24, 0, 65536);
    k_copych<<<dim3(256, 12, BATCH), 256, 0, stream>>>(E2, CAT2, 12, 65536, 24, 12);
    CONV(24,12, 256, 256, CAT2, TMP, 6); do_bn(TMP, 6, 0, 65536);       // dec2 c1
    CONV(12,12, 256, 256, TMP,  D2,  7); do_bn(D2,  7, 1, 65536);       // dec2 c2

    k_upconv<12><<<dim3(1024, 6, BATCH), 256, 0, stream>>>(              // up1 -> CAT1[0:6)
        D2, (const float*)d_in[27], (const float*)d_in[28], CAT1, 6, 9, 256, 65536, 12, 0, 262144);
    k_copych<<<dim3(1024, 6, BATCH), 256, 0, stream>>>(E1, CAT1, 6, 262144, 12, 6);
    CONV(12, 6, 512, 512, CAT1, TMP, 8); do_bn(TMP, 8, 0, 262144);      // dec1 c1
    CONV(6,  6, 512, 512, TMP,  D1,  9); do_bn(D1,  9, 1, 262144);      // dec1 c2

    // ---- pix = [out_conv(d1), x, coords] ----
    k_buildpix<<<dim3(1024, 8, BATCH), 256, 0, stream>>>(
        D1, (const float*)d_in[35], (const float*)d_in[36], x, PIX);

    // ---- SLIC ----
    k_initc<<<dim3(16, 8, BATCH), 256, 0, stream>>>(PIX, CEN0);
    float* cur = CEN0; float* nxt = CEN1;
    for (int it = 0; it < 5; ++it) {
        k_assign<<<dim3(64, 64, BATCH), 64, 0, stream>>>(PIX, cur, SIMS);
        if (it < 4) {
            int t = BATCH * 64 * 64;
            k_update<<<(t + 255) / 256, 256, 0, stream>>>(PIX, SIMS, nxt, t);
            float* sw = cur; cur = nxt; nxt = sw;
        }
    }
#undef CONV
}
